// SSDLinearAttention_74388833566901
// MI455X (gfx1250) — compile-verified, matrix operand order checked by reading, not run on hardware
//
#include <hip/hip_runtime.h>
#include <hip/hip_bf16.h>

// ---------------------------------------------------------------------------
// SSD linear attention for MI455X (gfx1250), chunked formulation, bf16 WMMA.
// GEMMs: double-buffered LDS with GLOBAL_LOAD_ASYNC_TO_LDS_B128 staging
// (ASYNCcnt), v_wmma_f32_16x16x32_bf16 compute, fp32 accumulate.
// ---------------------------------------------------------------------------

typedef __attribute__((ext_vector_type(16))) __bf16       v16bf;
typedef __attribute__((ext_vector_type(8)))  float        v8f;
typedef __attribute__((ext_vector_type(4)))  unsigned int u32x4;

#define WMMA_BF16(a, b, c) \
  __builtin_amdgcn_wmma_f32_16x16x32_bf16(false, (a), false, (b), (short)0, (c), false, false)

static constexpr int BB = 2, NN = 8192, DD = 1024, HH = 16, DH = 64, PP = 1024;
static constexpr int LC = 64;               // chunk length
static constexpr int NCH = NN / LC;         // 128 chunks per sequence
static constexpr int STATE = DH * DH + DH;  // 4160 floats: S (64x64) + z (64)
static constexpr float EPSV = 1e-6f;
static constexpr float DECAY_SHIFT = 2.0f;

__device__ inline v8f zero8() {
  v8f z;
#pragma unroll
  for (int i = 0; i < 8; ++i) z[i] = 0.f;
  return z;
}

// Workgroup-relative LDS byte address (async-to-LDS VDST wants LDS_BASE-relative).
__device__ inline unsigned lds_addr(const void* p) {
  return (unsigned)(unsigned long long)(const __attribute__((address_space(3))) void*)p;
}

// Load one 16x32 bf16 operand fragment (A layout; B uses same layout on a
// transposed tile). Tile row-major in LDS, row stride ldr (mult of 8 -> rows
// 16B aligned). Lane l: row = l&15; kb = (l&16)?8:0; elems 0..7 <- [kb..kb+7],
// elems 8..15 <- [16+kb..16+kb+7]  (CDNA5 ISA 16-bit A-matrix layout).
__device__ inline v16bf frag_ld(const __bf16* tile, int row0, int ldr, int koff) {
  int lane = threadIdx.x & 31;
  int r = lane & 15;
  int kb = (lane & 16) ? 8 : 0;
  const __bf16* p = tile + (size_t)(row0 + r) * ldr + koff + kb;
  union { v16bf v; u32x4 u[2]; } f;
  f.u[0] = *(const u32x4*)(p);
  f.u[1] = *(const u32x4*)(p + 16);
  return f.v;
}

__device__ inline void acc_coords(int lane, int g, int& m, int& n) {
  m = g + ((lane & 16) ? 8 : 0);  // C/D: VGPR g holds row m (or m+8)
  n = lane & 15;                  // column striped across lanes
}

// Issue async copies for one 128x32 bf16 tile (row-major src, stride K elems)
// into LDS at ldsbase with padded row stride 40. 256 threads x 2 x 16B = 8KB.
__device__ __forceinline__ void stage_async_tile(const __bf16* __restrict__ src,
                                                 unsigned ldsbase, int k0, int K,
                                                 int tid) {
#pragma unroll
  for (int it = 0; it < 2; ++it) {
    int id = tid + it * 256;          // 0..511
    int row = id >> 2;                // 0..127
    int cg = (id & 3) * 8;            // k-chunk of 8 bf16 = 16B
    const __bf16* g = src + (size_t)row * K + k0 + cg;
    unsigned l = ldsbase + (unsigned)(row * 40 + cg) * 2u;
    asm volatile("global_load_async_to_lds_b128 %0, %1, off"
                 :: "v"(l), "v"(g) : "memory");
  }
}

__device__ __forceinline__ void wait_async0() {
  asm volatile("s_wait_asynccnt 0x0" ::: "memory");
}

// ---------------------------------------------------------------------------
// Kernel 0a: fp32 -> bf16 bulk convert (vectorized 8/thread).
// ---------------------------------------------------------------------------
__global__ __launch_bounds__(256) void to_bf16(const float* __restrict__ src,
                                               __bf16* __restrict__ dst) {
  size_t i = ((size_t)blockIdx.x * 256 + threadIdx.x) * 8;
#pragma unroll
  for (int j = 0; j < 8; ++j) dst[i + j] = (__bf16)src[i + j];
}

// ---------------------------------------------------------------------------
// Kernel 0b: W[R][C] fp32 -> WT[C][R] bf16 (32x32 LDS tiles, 256 threads).
// ---------------------------------------------------------------------------
__global__ __launch_bounds__(256) void transpose_bf16(
    const float* __restrict__ W, __bf16* __restrict__ WT, int R, int Ccols) {
  __shared__ float tile[32][33];
  int r0 = blockIdx.y * 32, c0 = blockIdx.x * 32;
  int tx = threadIdx.x & 31, ty = threadIdx.x >> 5;  // 32 x 8
#pragma unroll
  for (int i = 0; i < 32; i += 8)
    tile[ty + i][tx] = W[(size_t)(r0 + ty + i) * Ccols + c0 + tx];
  __syncthreads();
#pragma unroll
  for (int i = 0; i < 32; i += 8)
    WT[(size_t)(c0 + ty + i) * R + r0 + tx] = (__bf16)tile[tx][ty + i];
}

// ---------------------------------------------------------------------------
// Kernel 1: bf16 WMMA GEMM  C[M,N] = act(A[M,K] @ BT[N,K]^T).
// A, BT bf16 row-major (BT is pre-transposed B). Block 128x128, K-step 32,
// 256 threads = 8 waves, wave tile 32x64 (2x4 WMMA accums).
// Double-buffered LDS; tiles staged with GLOBAL_LOAD_ASYNC_TO_LDS_B128 so
// global latency overlaps the WMMA pipe; ASYNCcnt fenced before each barrier.
// ACT: 0 identity, 1 elu(x)+1.  OT: float or __bf16 output.
// ---------------------------------------------------------------------------
template <int ACT, typename OT>
__global__ __launch_bounds__(256) void gemm_bf16(
    const __bf16* __restrict__ A, const __bf16* __restrict__ BT,
    OT* __restrict__ C, int M, int N, int K) {
  constexpr int LDT = 40;
  __shared__ __align__(16) __bf16 sA[2][128 * LDT];
  __shared__ __align__(16) __bf16 sB[2][128 * LDT];

  const int tid = threadIdx.x;
  const int lane = tid & 31;
  const int wave = tid >> 5;
  const int wm = wave >> 1;  // 0..3
  const int wn = wave & 1;   // 0..1
  const int m0 = blockIdx.y * 128;
  const int n0 = blockIdx.x * 128;
  const __bf16* Ab = A + (size_t)m0 * K;
  const __bf16* Bb = BT + (size_t)n0 * K;
  unsigned lA[2] = {lds_addr(&sA[0][0]), lds_addr(&sA[1][0])};
  unsigned lB[2] = {lds_addr(&sB[0][0]), lds_addr(&sB[1][0])};

  v8f acc[2][4];
#pragma unroll
  for (int i = 0; i < 2; ++i)
#pragma unroll
    for (int j = 0; j < 4; ++j) acc[i][j] = zero8();

  stage_async_tile(Ab, lA[0], 0, K, tid);
  stage_async_tile(Bb, lB[0], 0, K, tid);
  wait_async0();
  __syncthreads();

  const int NK = K >> 5;
  for (int kt = 0; kt < NK; ++kt) {
    const int cur = kt & 1;
    if (kt + 1 < NK) {  // prefetch next tiles into the other buffer
      stage_async_tile(Ab, lA[cur ^ 1], (kt + 1) * 32, K, tid);
      stage_async_tile(Bb, lB[cur ^ 1], (kt + 1) * 32, K, tid);
    }
    v16bf afr[2], bfr[4];
#pragma unroll
    for (int i = 0; i < 2; ++i) afr[i] = frag_ld(sA[cur], wm * 32 + i * 16, LDT, 0);
#pragma unroll
    for (int j = 0; j < 4; ++j) bfr[j] = frag_ld(sB[cur], wn * 64 + j * 16, LDT, 0);
#pragma unroll
    for (int i = 0; i < 2; ++i)
#pragma unroll
      for (int j = 0; j < 4; ++j)
        acc[i][j] = WMMA_BF16(afr[i], bfr[j], acc[i][j]);
    if (kt + 1 < NK) wait_async0();  // next buffer fully written
    __syncthreads();                 // all waves done reading cur buffer
  }

#pragma unroll
  for (int i = 0; i < 2; ++i)
#pragma unroll
    for (int j = 0; j < 4; ++j)
#pragma unroll
      for (int g = 0; g < 8; ++g) {
        int m, n;
        acc_coords(lane, g, m, n);
        int gm = m0 + wm * 32 + i * 16 + m;
        int gn = n0 + wn * 64 + j * 16 + n;
        float v = acc[i][j][g];
        if (ACT == 1) v = (v > 0.f) ? (v + 1.f) : __expf(v);  // elu(x)+1
        C[(size_t)gm * N + gn] = (OT)v;
      }
}

// ---------------------------------------------------------------------------
// Kernel 2: decay gate a = sigmoid(x @ Wa + ba + 2).  (B,N,H) fp32 output.
// ---------------------------------------------------------------------------
__global__ __launch_bounds__(128) void proj_a(
    const float* __restrict__ x, const float* __restrict__ Wa,
    const float* __restrict__ ba, float* __restrict__ abuf) {
  int tok = blockIdx.x * 8 + (threadIdx.x >> 4);
  int h = threadIdx.x & 15;
  const float* xr = x + (size_t)tok * DD;
  float s = 0.f;
#pragma unroll 4
  for (int j = 0; j < DD; ++j) s += xr[j] * Wa[j * HH + h];
  s += ba[h] + DECAY_SHIFT;
  abuf[(size_t)tok * HH + h] = 1.f / (1.f + __expf(-s));
}

// ---------------------------------------------------------------------------
// Kernel 3: per-chunk local state  S_loc = sum_t g_t k_t v_t^T
// (g_t = prod of decays strictly after t), z_loc = sum_t g_t k_t, chunk
// decay A_c. One block per (b,h,chunk): 64x64x64 bf16 WMMA GEMM.
// ---------------------------------------------------------------------------
__global__ __launch_bounds__(128) void chunk_state(
    const __bf16* __restrict__ kbuf, const __bf16* __restrict__ vbuf,
    const float* __restrict__ abuf, float* __restrict__ Sbuf,
    float* __restrict__ Avals) {
  constexpr int LDT = 72;
  const int cid = blockIdx.x;
  const int c = cid % NCH;
  const int h = (cid / NCH) % HH;
  const int b = cid / (NCH * HH);
  const int n0 = c * LC;

  __shared__ __align__(16) __bf16 sKT[DH * LDT];  // [d][t]  (A: rows=d, K=t)
  __shared__ __align__(16) __bf16 sVT[DH * LDT];  // [d'][t] (B^T)
  __shared__ float g[LC], av[LC];

  const int tid = threadIdx.x;
  const int lane = tid & 31;
  const int wave = tid >> 5;

  if (tid < LC) av[tid] = abuf[((size_t)b * NN + n0 + tid) * HH + h];
  __syncthreads();
  if (tid == 0) {
    float gg = 1.f;
    for (int t = LC - 1; t >= 0; --t) { g[t] = gg; gg *= av[t]; }
    Avals[cid] = gg;
  }
  __syncthreads();

  const size_t kvbase = ((size_t)b * NN + n0) * PP + (size_t)h * DH;
#pragma unroll
  for (int it = 0; it < 32; ++it) {
    int idx = tid + it * 128;
    int t = idx >> 6, d = idx & 63;
    sKT[d * LDT + t] = (__bf16)((float)kbuf[kvbase + (size_t)t * PP + d] * g[t]);
    sVT[d * LDT + t] = vbuf[kvbase + (size_t)t * PP + d];
  }
  __syncthreads();

  v8f acc[4];
#pragma unroll
  for (int j = 0; j < 4; ++j) acc[j] = zero8();
#pragma unroll
  for (int kt = 0; kt < 2; ++kt) {
    v16bf af = frag_ld(sKT, wave * 16, LDT, kt * 32);
#pragma unroll
    for (int j = 0; j < 4; ++j) {
      v16bf bf = frag_ld(sVT, j * 16, LDT, kt * 32);
      acc[j] = WMMA_BF16(af, bf, acc[j]);
    }
  }

  float* Sc = Sbuf + (size_t)cid * STATE;
#pragma unroll
  for (int j = 0; j < 4; ++j)
#pragma unroll
    for (int gg = 0; gg < 8; ++gg) {
      int m, n;
      acc_coords(lane, gg, m, n);
      Sc[(wave * 16 + m) * DH + j * 16 + n] = acc[j][gg];
    }
  __syncthreads();
  if (tid < DH) {  // z_loc[d]
    float z = 0.f;
#pragma unroll 8
    for (int t = 0; t < LC; ++t) z += (float)sKT[tid * LDT + t];
    Sc[DH * DH + tid] = z;
  }
}

// ---------------------------------------------------------------------------
// Kernel 4: sequential scan over 128 chunk states per (b,h), in place:
// reads local state, writes the state entering the chunk. z rides along.
// ---------------------------------------------------------------------------
__global__ __launch_bounds__(256) void chunk_scan(
    float* __restrict__ Sbuf, const float* __restrict__ Avals) {
  const int bh = blockIdx.x;
  const int tid = threadIdx.x;
  float S[17];
#pragma unroll
  for (int i = 0; i < 17; ++i) S[i] = 0.f;
  float* base = Sbuf + (size_t)bh * NCH * STATE;
  const float* Ab = Avals + (size_t)bh * NCH;
  for (int c = 0; c < NCH; ++c) {
    float* p = base + (size_t)c * STATE;
    float Ac = Ab[c];
#pragma unroll
    for (int i = 0; i < 17; ++i) {
      int idx = tid + i * 256;
      if (idx < STATE) {
        float loc = p[idx];
        p[idx] = S[i];
        S[i] = Ac * S[i] + loc;
      }
    }
  }
}

// ---------------------------------------------------------------------------
// Kernel 5: per-chunk output.
//   c_t = prod_{s<=t} a_s;  P = tril((c_t q)@(k/c_s)^T)
//   num = P@V + (c_t q)@S_init;  den = (c_t q).z_init + rowsum(P) + eps
//   attn[t][d] = num/den  (bf16, (B,N,P) layout)
// ---------------------------------------------------------------------------
__global__ __launch_bounds__(128) void chunk_output(
    const __bf16* __restrict__ qbuf, const __bf16* __restrict__ kbuf,
    const __bf16* __restrict__ vbuf, const float* __restrict__ abuf,
    const float* __restrict__ Sbuf, __bf16* __restrict__ attn) {
  constexpr int LDT = 72;
  const int cid = blockIdx.x;
  const int c = cid % NCH;
  const int h = (cid / NCH) % HH;
  const int b = cid / (NCH * HH);
  const int n0 = c * LC;

  __shared__ __align__(16) __bf16 sQ[LC * LDT];   // [t][d] * c_t
  __shared__ __align__(16) __bf16 sK[LC * LDT];   // [s][d] / c_s
  __shared__ __align__(16) __bf16 sVT[DH * LDT];  // [d][s]
  __shared__ __align__(16) __bf16 sST[DH * LDT];  // [d][d0] = S_init^T
  __shared__ __align__(16) __bf16 sP[LC * LDT];   // [t][s] masked
  __shared__ float cf[LC], cinv[LC], av[LC], zi[DH], invden[LC];

  const int tid = threadIdx.x;
  const int lane = tid & 31;
  const int wave = tid >> 5;

  if (tid < LC) av[tid] = abuf[((size_t)b * NN + n0 + tid) * HH + h];
  __syncthreads();
  if (tid == 0) {
    float cc = 1.f;
    for (int t = 0; t < LC; ++t) {
      cc *= av[t];
      cf[t] = cc;
      cinv[t] = 1.f / fmaxf(cc, 1e-30f);
    }
  }
  __syncthreads();

  const float* Sc = Sbuf + (size_t)cid * STATE;
  const size_t kvbase = ((size_t)b * NN + n0) * PP + (size_t)h * DH;
  if (tid < DH) zi[tid] = Sc[DH * DH + tid];
#pragma unroll
  for (int it = 0; it < 32; ++it) {
    int idx = tid + it * 128;
    int t = idx >> 6, d = idx & 63;
    sQ[t * LDT + d] = (__bf16)((float)qbuf[kvbase + (size_t)t * PP + d] * cf[t]);
    sK[t * LDT + d] = (__bf16)((float)kbuf[kvbase + (size_t)t * PP + d] * cinv[t]);
    sVT[d * LDT + t] = vbuf[kvbase + (size_t)t * PP + d];
    sST[d * LDT + t] = (__bf16)Sc[t * DH + d];  // t plays d0 here
  }
  __syncthreads();

  // P = sQ @ sK^T
  v8f pacc[4];
#pragma unroll
  for (int j = 0; j < 4; ++j) pacc[j] = zero8();
#pragma unroll
  for (int kt = 0; kt < 2; ++kt) {
    v16bf af = frag_ld(sQ, wave * 16, LDT, kt * 32);
#pragma unroll
    for (int j = 0; j < 4; ++j) {
      v16bf bf = frag_ld(sK, j * 16, LDT, kt * 32);
      pacc[j] = WMMA_BF16(af, bf, pacc[j]);
    }
  }
  // causal mask in accumulators, spill masked P (bf16)
#pragma unroll
  for (int j = 0; j < 4; ++j)
#pragma unroll
    for (int g = 0; g < 8; ++g) {
      int m, n;
      acc_coords(lane, g, m, n);
      int t = wave * 16 + m, s = j * 16 + n;
      sP[t * LDT + s] = (__bf16)((s <= t) ? pacc[j][g] : 0.f);
    }
  __syncthreads();

  if (tid < LC) {  // rowsum + denominator per token
    float r = 0.f;
#pragma unroll 8
    for (int s = 0; s < LC; ++s) r += (float)sP[tid * LDT + s];
    float dq = 0.f;
#pragma unroll 8
    for (int d = 0; d < DH; ++d) dq += (float)sQ[tid * LDT + d] * zi[d];
    invden[tid] = 1.f / (dq + r + EPSV);
  }

  // num = P @ V + sQ @ S_init
  v8f acc[4];
#pragma unroll
  for (int j = 0; j < 4; ++j) acc[j] = zero8();
#pragma unroll
  for (int kt = 0; kt < 2; ++kt) {
    v16bf ap = frag_ld(sP, wave * 16, LDT, kt * 32);
    v16bf aq = frag_ld(sQ, wave * 16, LDT, kt * 32);
#pragma unroll
    for (int j = 0; j < 4; ++j) {
      v16bf bv = frag_ld(sVT, j * 16, LDT, kt * 32);
      acc[j] = WMMA_BF16(ap, bv, acc[j]);
      v16bf bs = frag_ld(sST, j * 16, LDT, kt * 32);
      acc[j] = WMMA_BF16(aq, bs, acc[j]);
    }
  }
  __syncthreads();

#pragma unroll
  for (int j = 0; j < 4; ++j)
#pragma unroll
    for (int g = 0; g < 8; ++g) {
      int m, n;
      acc_coords(lane, g, m, n);
      int t = wave * 16 + m, d = j * 16 + n;
      attn[kvbase + (size_t)t * PP + d] = (__bf16)(acc[j][g] * invden[t]);
    }
}

// ---------------------------------------------------------------------------
// Host-side launcher (graph-capture safe).
// ---------------------------------------------------------------------------
extern "C" void kernel_launch(void* const* d_in, const int* in_sizes, int n_in,
                              void* d_out, int out_size, void* d_ws, size_t ws_size,
                              hipStream_t stream) {
  (void)in_sizes; (void)n_in; (void)out_size; (void)ws_size;
  const float* x  = (const float*)d_in[0];
  const float* Wq = (const float*)d_in[1];
  const float* Wk = (const float*)d_in[2];
  const float* Wv = (const float*)d_in[3];
  const float* Wo = (const float*)d_in[4];
  const float* Wa = (const float*)d_in[5];
  const float* ba = (const float*)d_in[6];
  float* out = (float*)d_out;

  char* base = (char*)d_ws;
  size_t off = 0;
  auto alloc = [&](size_t bytes) {
    void* p = base + off;
    off = (off + bytes + 255) & ~(size_t)255;
    return p;
  };
  const size_t MT = (size_t)BB * NN;  // 16384 tokens
  __bf16* xbf  = (__bf16*)alloc(MT * DD * 2);
  __bf16* WqT  = (__bf16*)alloc((size_t)DD * PP * 2);
  __bf16* WkT  = (__bf16*)alloc((size_t)DD * PP * 2);
  __bf16* WvT  = (__bf16*)alloc((size_t)DD * PP * 2);
  __bf16* WoT  = (__bf16*)alloc((size_t)PP * DD * 2);
  __bf16* qbf  = (__bf16*)alloc(MT * PP * 2);
  __bf16* kbf  = (__bf16*)alloc(MT * PP * 2);
  __bf16* vbf  = (__bf16*)alloc(MT * PP * 2);
  __bf16* attn = (__bf16*)alloc(MT * PP * 2);
  float* abuf  = (float*)alloc(MT * HH * 4);
  float* Avls  = (float*)alloc((size_t)BB * HH * NCH * 4);
  float* Sbuf  = (float*)alloc((size_t)BB * HH * NCH * STATE * 4);

  const int M = (int)MT;
  dim3 blk(256);

  to_bf16<<<(unsigned)(MT * DD / 2048), blk, 0, stream>>>(x, xbf);
  dim3 gT(PP / 32, DD / 32);
  transpose_bf16<<<gT, blk, 0, stream>>>(Wq, WqT, DD, PP);
  transpose_bf16<<<gT, blk, 0, stream>>>(Wk, WkT, DD, PP);
  transpose_bf16<<<gT, blk, 0, stream>>>(Wv, WvT, DD, PP);
  dim3 gTo(DD / 32, PP / 32);
  transpose_bf16<<<gTo, blk, 0, stream>>>(Wo, WoT, PP, DD);

  dim3 gQKV(PP / 128, M / 128);  // (8,128)
  hipLaunchKernelGGL(HIP_KERNEL_NAME(gemm_bf16<1, __bf16>), gQKV, blk, 0, stream,
                     xbf, WqT, qbf, M, PP, DD);
  hipLaunchKernelGGL(HIP_KERNEL_NAME(gemm_bf16<1, __bf16>), gQKV, blk, 0, stream,
                     xbf, WkT, kbf, M, PP, DD);
  hipLaunchKernelGGL(HIP_KERNEL_NAME(gemm_bf16<0, __bf16>), gQKV, blk, 0, stream,
                     xbf, WvT, vbf, M, PP, DD);
  proj_a<<<M / 8, 128, 0, stream>>>(x, Wa, ba, abuf);

  chunk_state<<<BB * HH * NCH, 128, 0, stream>>>(kbf, vbf, abuf, Sbuf, Avls);
  chunk_scan<<<BB * HH, 256, 0, stream>>>(Sbuf, Avls);
  chunk_output<<<BB * HH * NCH, 128, 0, stream>>>(qbf, kbf, vbf, abuf, Sbuf, attn);

  dim3 gOut(DD / 128, M / 128);
  hipLaunchKernelGGL(HIP_KERNEL_NAME(gemm_bf16<0, float>), gOut, blk, 0, stream,
                     attn, WoT, out, M, DD, PP);
}
